// MultiHeadSelfAttention_60086592471644
// MI455X (gfx1250) — matrix operand order repaired and checked by reading, not compiled, not run on hardware
//
#include <hip/hip_runtime.h>
#include <hip/hip_bf16.h>

// ---------------------------------------------------------------------------
// MI455X (gfx1250) multi-head self-attention, bf16 WMMA pipeline:
//   x,W -> bf16 ; Q/K GEMM ; V GEMM (transposed out) ; fused flash attention ;
//   O GEMM (f32 out). All matmuls use v_wmma_f32_16x16x32_bf16 (wave32).
// ---------------------------------------------------------------------------

typedef __bf16 bf16_t;
typedef __attribute__((ext_vector_type(16))) __bf16 v16bf;
typedef __attribute__((ext_vector_type(8)))  float  v8f;
typedef __attribute__((ext_vector_type(4)))  unsigned int v4u;

#define D_MODEL 1024
#define NHEAD   16
#define DHEAD   64
#define SEQ     2048
#define BATCH   4
#define MTOT    (BATCH * SEQ)
#define LOG2E   1.44269504088896340736f

union FragCast { v16bf v; v4u u[2]; };

// A fragment (16x32 bf16, MxK): lane 0-15 -> M=lane, K 0-7 & 16-23;
// lane 16-31 -> M=lane-16, K 8-15 & 24-31.  (ISA 7.12.2)
static __device__ inline v16bf load_frag_a(const bf16_t* p, int ld, int row0, int k0) {
  int lane = threadIdx.x & 31;
  int half = lane >> 4;
  const bf16_t* base = p + (size_t)(row0 + (lane & 15)) * ld + (k0 + half * 8);
  FragCast r;
  r.u[0] = *(const v4u*)(base);
  r.u[1] = *(const v4u*)(base + 16);
  return r.v;
}

// B fragment (32x16 bf16, KxN) where logical B[k][n] = p[n*ld + k]
// (weights row-major, K row-major, Vt row-major all fit this form):
// lane 0-15 -> N=lane, K 0-7 & 16-23 ; lane 16-31 -> N=lane-16, K 8-15 & 24-31.
static __device__ inline v16bf load_frag_b(const bf16_t* p, int ld, int k0, int n0) {
  int lane = threadIdx.x & 31;
  int half = lane >> 4;
  const bf16_t* base = p + (size_t)(n0 + (lane & 15)) * ld + (k0 + half * 8);
  FragCast r;
  r.u[0] = *(const v4u*)(base);
  r.u[1] = *(const v4u*)(base + 16);
  return r.v;
}

static __device__ inline v8f wmma_bf16(v16bf a, v16bf b, v8f c) {
  return __builtin_amdgcn_wmma_f32_16x16x32_bf16(false, a, false, b, (short)0, c,
                                                 false, false);
}

// ---------------------------------------------------------------------------
__global__ void f32_to_bf16_kernel(const float* __restrict__ in,
                                   bf16_t* __restrict__ out, int n) {
  int i = blockIdx.x * blockDim.x + threadIdx.x;
  if (i < n) out[i] = (bf16_t)in[i];
}

// ---------------------------------------------------------------------------
// C[M,N] = A[M,K] * W[N,K]^T.  Wave computes 64x64 tile; block = 4 waves =
// 128x128 tile.  MODE 0: bf16 row-major out. MODE 1: bf16 transposed out
// (Vt[b][n][s], per-batch). MODE 2: f32 row-major out.
template <int MODE>
__global__ __launch_bounds__(128) void gemm_wmma_kernel(
    const bf16_t* __restrict__ A, const bf16_t* __restrict__ W,
    void* __restrict__ Cout, int M, int N, int K) {
  int wave = threadIdx.x >> 5;
  int m0 = blockIdx.x * 128 + (wave & 1) * 64;
  int n0 = blockIdx.y * 128 + (wave >> 1) * 64;

  v8f acc[4][4] = {};
  for (int k0 = 0; k0 < K; k0 += 32) {
    v16bf a[4], b[4];
#pragma unroll
    for (int i = 0; i < 4; ++i) a[i] = load_frag_a(A, K, m0 + 16 * i, k0);
#pragma unroll
    for (int j = 0; j < 4; ++j) b[j] = load_frag_b(W, K, k0, n0 + 16 * j);
#pragma unroll
    for (int i = 0; i < 4; ++i)
#pragma unroll
      for (int j = 0; j < 4; ++j) acc[i][j] = wmma_bf16(a[i], b[j], acc[i][j]);
  }

  int lane = threadIdx.x & 31;
  int half = lane >> 4;
  int col  = lane & 15;
#pragma unroll
  for (int i = 0; i < 4; ++i)
#pragma unroll
    for (int j = 0; j < 4; ++j)
#pragma unroll
      for (int r = 0; r < 8; ++r) {
        int row = m0 + 16 * i + 8 * half + r;
        int c   = n0 + 16 * j + col;
        float v = acc[i][j][r];
        if (MODE == 0) {
          ((bf16_t*)Cout)[(size_t)row * N + c] = (bf16_t)v;
        } else if (MODE == 1) {
          int bb = row >> 11;            // row / SEQ
          int ml = row & (SEQ - 1);      // row % SEQ
          ((bf16_t*)Cout)[((size_t)bb * N + c) * SEQ + ml] = (bf16_t)v;
        } else {
          ((float*)Cout)[(size_t)row * N + c] = v;
        }
      }
}

// ---------------------------------------------------------------------------
// Fused flash attention. Wave handles 32 query rows of one (b,h).
// Q,K: [b][s][1024] bf16 (head slice cols h*64..).  Vt: [b][1024][s] bf16.
// Ctx out: [b][s][1024] bf16.
__global__ __launch_bounds__(128) void attn_wmma_kernel(
    const bf16_t* __restrict__ Q, const bf16_t* __restrict__ Km,
    const bf16_t* __restrict__ Vt, bf16_t* __restrict__ Ctx) {
  __shared__ __align__(16) bf16_t pls[4][32 * 64];  // per-wave P tile

  int wave = threadIdx.x >> 5;
  int lane = threadIdx.x & 31;
  int half = lane >> 4;
  int col  = lane & 15;
  int b = blockIdx.z, h = blockIdx.y;
  int q0 = (blockIdx.x * 4 + wave) * 32;

  const bf16_t* Qb = Q  + (size_t)b * SEQ * D_MODEL + h * DHEAD;
  const bf16_t* Kb = Km + (size_t)b * SEQ * D_MODEL + h * DHEAD;
  const bf16_t* Vb = Vt + ((size_t)b * D_MODEL + h * DHEAD) * SEQ;
  bf16_t* lds = pls[wave];

  // preload Q fragments (A layout), d_head = 64 -> 2 k-steps
  v16bf qa[2][2];
#pragma unroll
  for (int mt = 0; mt < 2; ++mt)
#pragma unroll
    for (int ks = 0; ks < 2; ++ks)
      qa[mt][ks] = load_frag_a(Qb, D_MODEL, q0 + 16 * mt, 32 * ks);

  float m_run[2][8], l_run[2][8];
  v8f o[2][4] = {};
#pragma unroll
  for (int mt = 0; mt < 2; ++mt)
#pragma unroll
    for (int r = 0; r < 8; ++r) { m_run[mt][r] = -1e30f; l_run[mt][r] = 0.f; }

  for (int k0 = 0; k0 < SEQ; k0 += 64) {
    // ---- S = (Q K^T) / sqrt(d_head), 32x64 tile ----
    v8f s[2][4] = {};
#pragma unroll
    for (int ks = 0; ks < 2; ++ks) {
      v16bf kb[4];
#pragma unroll
      for (int jt = 0; jt < 4; ++jt)
        kb[jt] = load_frag_b(Kb, D_MODEL, 32 * ks, k0 + 16 * jt);
#pragma unroll
      for (int mt = 0; mt < 2; ++mt)
#pragma unroll
        for (int jt = 0; jt < 4; ++jt)
          s[mt][jt] = wmma_bf16(qa[mt][ks], kb[jt], s[mt][jt]);
    }
#pragma unroll
    for (int mt = 0; mt < 2; ++mt)
#pragma unroll
      for (int jt = 0; jt < 4; ++jt)
#pragma unroll
        for (int r = 0; r < 8; ++r) s[mt][jt][r] *= 0.125f;  // 1/sqrt(64)

    // ---- online softmax update ----
#pragma unroll
    for (int mt = 0; mt < 2; ++mt) {
#pragma unroll
      for (int r = 0; r < 8; ++r) {
        float mx = -1e30f;
#pragma unroll
        for (int jt = 0; jt < 4; ++jt) mx = fmaxf(mx, s[mt][jt][r]);
#pragma unroll
        for (int off = 8; off >= 1; off >>= 1)       // stays within 16-lane half
          mx = fmaxf(mx, __shfl_xor(mx, off, 32));
        float mnew  = fmaxf(m_run[mt][r], mx);
        float alpha = __builtin_amdgcn_exp2f((m_run[mt][r] - mnew) * LOG2E);
        m_run[mt][r] = mnew;
        float rowsum = 0.f;
#pragma unroll
        for (int jt = 0; jt < 4; ++jt) {
          float p = __builtin_amdgcn_exp2f((s[mt][jt][r] - mnew) * LOG2E);
          s[mt][jt][r] = p;
          rowsum += p;
        }
#pragma unroll
        for (int off = 8; off >= 1; off >>= 1)
          rowsum += __shfl_xor(rowsum, off, 32);
        l_run[mt][r] = l_run[mt][r] * alpha + rowsum;
#pragma unroll
        for (int nt = 0; nt < 4; ++nt) o[mt][nt][r] *= alpha;
      }
    }

    // ---- P (C layout) -> LDS -> A layout ----
#pragma unroll
    for (int mt = 0; mt < 2; ++mt)
#pragma unroll
      for (int jt = 0; jt < 4; ++jt)
#pragma unroll
        for (int r = 0; r < 8; ++r)
          lds[(mt * 16 + 8 * half + r) * 64 + jt * 16 + col] =
              (bf16_t)s[mt][jt][r];

    // ---- O += P @ V ----
#pragma unroll
    for (int ks = 0; ks < 2; ++ks) {
      v16bf pa[2];
#pragma unroll
      for (int mt = 0; mt < 2; ++mt)
        pa[mt] = load_frag_a(lds, 64, mt * 16, 32 * ks);
      v16bf vb[4];
#pragma unroll
      for (int nt = 0; nt < 4; ++nt)
        vb[nt] = load_frag_b(Vb, SEQ, k0 + 32 * ks, nt * 16);
#pragma unroll
      for (int mt = 0; mt < 2; ++mt)
#pragma unroll
        for (int nt = 0; nt < 4; ++nt)
          o[mt][nt] = wmma_bf16(pa[mt], vb[nt], o[mt][nt]);
    }
  }

  // ---- normalize and store ctx ----
#pragma unroll
  for (int mt = 0; mt < 2; ++mt)
#pragma unroll
    for (int r = 0; r < 8; ++r) {
      float inv = 1.0f / l_run[mt][r];
      int row = q0 + mt * 16 + 8 * half + r;
#pragma unroll
      for (int nt = 0; nt < 4; ++nt) {
        int d = nt * 16 + col;
        Ctx[((size_t)b * SEQ + row) * D_MODEL + h * DHEAD + d] =
            (bf16_t)(o[mt][nt][r] * inv);
      }
    }
}

// ---------------------------------------------------------------------------
extern "C" void kernel_launch(void* const* d_in, const int* in_sizes, int n_in,
                              void* d_out, int out_size, void* d_ws, size_t ws_size,
                              hipStream_t stream) {
  const float* x  = (const float*)d_in[0];
  const float* wq = (const float*)d_in[1];
  const float* wk = (const float*)d_in[2];
  const float* wv = (const float*)d_in[3];
  const float* wo = (const float*)d_in[4];

  char* ws = (char*)d_ws;
  size_t off = 0;
  const size_t XB = (size_t)MTOT * D_MODEL * sizeof(bf16_t);      // 16.78 MB
  const size_t WB = (size_t)D_MODEL * D_MODEL * sizeof(bf16_t);   //  2.10 MB
  bf16_t* Xb  = (bf16_t*)(ws + off); off += XB;
  bf16_t* Wqb = (bf16_t*)(ws + off); off += WB;
  bf16_t* Wkb = (bf16_t*)(ws + off); off += WB;
  bf16_t* Wvb = (bf16_t*)(ws + off); off += WB;
  bf16_t* Wob = (bf16_t*)(ws + off); off += WB;
  bf16_t* Qb  = (bf16_t*)(ws + off); off += XB;
  bf16_t* Kb  = (bf16_t*)(ws + off); off += XB;
  bf16_t* Vtb = (bf16_t*)(ws + off); off += XB;   // [b][n][s]
  bf16_t* Ctx = (bf16_t*)(ws + off); off += XB;

  const int NX = MTOT * D_MODEL;            // 8,388,608
  const int NW = D_MODEL * D_MODEL;         // 1,048,576
  f32_to_bf16_kernel<<<NX / 256, 256, 0, stream>>>(x,  Xb,  NX);
  f32_to_bf16_kernel<<<NW / 256, 256, 0, stream>>>(wq, Wqb, NW);
  f32_to_bf16_kernel<<<NW / 256, 256, 0, stream>>>(wk, Wkb, NW);
  f32_to_bf16_kernel<<<NW / 256, 256, 0, stream>>>(wv, Wvb, NW);
  f32_to_bf16_kernel<<<NW / 256, 256, 0, stream>>>(wo, Wob, NW);

  dim3 gg(MTOT / 128, D_MODEL / 128);       // 64 x 8
  gemm_wmma_kernel<0><<<gg, 128, 0, stream>>>(Xb, Wqb, Qb,  MTOT, D_MODEL, D_MODEL);
  gemm_wmma_kernel<0><<<gg, 128, 0, stream>>>(Xb, Wkb, Kb,  MTOT, D_MODEL, D_MODEL);
  gemm_wmma_kernel<1><<<gg, 128, 0, stream>>>(Xb, Wvb, Vtb, MTOT, D_MODEL, D_MODEL);

  attn_wmma_kernel<<<dim3(SEQ / 128, NHEAD, BATCH), 128, 0, stream>>>(Qb, Kb, Vtb, Ctx);

  gemm_wmma_kernel<2><<<gg, 128, 0, stream>>>(Ctx, Wob, d_out, MTOT, D_MODEL, D_MODEL);
}
